// HopfieldNetwork_58823872086839
// MI455X (gfx1250) — compile-verified, matrix operand order checked by reading, not run on hardware
//
#include <hip/hip_runtime.h>

// ---------------------------------------------------------------------------
// Hopfield network + STDP for MI455X (gfx1250, wave32, WMMA).
// Sequential chain (batch -> iter -> neuron-block) run by ONE persistent
// workgroup; state in LDS (f16, WMMA-A layout), W f16 mirror in L2-resident
// workspace. Both the neuron-potential sweep AND the energy form run on
// v_wmma_f32_16x16x32_f16; only STDP stays f32 scalar (learning precision).
// Energy is K-outer / block-inner with 7 accumulators per wave: the shared
// broadcast-s A fragment is loaded once per K-chunk (7x fewer LDS reads) and
// nothing is loop-invariant to hoist -> bounded VGPR pressure (<256, no
// s_set_vgpr_msb, >= 2 waves/SIMD). W mirror padded to 56 zero blocks so all
// waves run a uniform 7-block schedule (EXEC all-1s, no divergence).
// ---------------------------------------------------------------------------

typedef __attribute__((ext_vector_type(16))) _Float16 v16h;
typedef __attribute__((ext_vector_type(8)))  _Float16 v8h;
typedef __attribute__((ext_vector_type(8)))  float    v8f;

#define TSTEPS   8
#define NNEUR    784
#define NPAD     800          // 784 padded to multiple of 32 (K) -> zero tail
#define KCHUNKS  25           // NPAD / 32
#define NBLK     49           // NNEUR / 16 real neuron blocks
#define NBLKPAD  56           // 7 blocks/wave * 8 waves (zero-padded)
#define NROWPAD  (NBLKPAD * 16)   // 896 rows in padded f16 mirror
#define BPW      7            // blocks per wave in energy
#define BATCH    64
#define THREADS  256
#define WAVES    8

__device__ __forceinline__ unsigned hash_u32(unsigned x) {
    x ^= x >> 16; x *= 0x7feb352dU;
    x ^= x >> 15; x *= 0x846ca68bU;
    x ^= x >> 16; return x;
}
__device__ __forceinline__ float u01(unsigned x) {
    return (float)(x >> 8) * (1.0f / 16777216.0f);
}

// Build padded f16 mirror Wh[896][800] (zeros outside 784x784) and f32
// master copy Wf[784][784].
__global__ void hopfield_prep_kernel(const float* __restrict__ W,
                                     float* __restrict__ Wf,
                                     _Float16* __restrict__ Wh) {
    int idx = blockIdx.x * blockDim.x + threadIdx.x;
    if (idx >= NROWPAD * NPAD) return;
    int r = idx / NPAD, c = idx - r * NPAD;
    float w = 0.0f;
    if (r < NNEUR && c < NNEUR) { w = W[r * NNEUR + c]; Wf[r * NNEUR + c] = w; }
    Wh[idx] = (_Float16)w;
}

// Assemble a 16x32 f16 WMMA fragment from two contiguous 16B pieces.
__device__ __forceinline__ v16h frag16(const _Float16* base, int kb, int hiOff) {
    v8h lo = *(const v8h*)(base + kb + hiOff);
    v8h hi = *(const v8h*)(base + kb + 16 + hiOff);
    v16h f;
#pragma unroll
    for (int i = 0; i < 8; ++i) { f[i] = lo[i]; f[8 + i] = hi[i]; }
    return f;
}

// Block-wide energy on the WMMA pipe: q = W*s via 16x16x32 f16 (s is integer
// spike counts 0..8, exact in f16), then e = -0.5*dot(s,q) + sum(s).
// K-outer / block-inner: one shared A fragment feeds 7 WMMAs per K-chunk.
// Returns the same value to all threads.
__device__ float block_energy_wmma(const _Float16* __restrict__ Wh,
                                   const _Float16* s16, const float* s_sum,
                                   float* redA, float* redB,
                                   int tid, int wave, int lane,
                                   int nloc, int hiOff) {
    v8f zero = {};
    v8f acc[BPW];
#pragma unroll
    for (int i = 0; i < BPW; ++i) acc[i] = zero;

    // One base pointer; per-block stride is a compile-time constant that
    // folds into the 24-bit instruction offset (no per-block address VALU).
    const _Float16* wbase = Wh + (size_t)(wave * 16 + nloc) * NPAD;

    for (int kc = 0; kc < KCHUNKS; ++kc) {
        int kb = kc * 32;
        v16h a = frag16(s16, kb, hiOff);   // shared A: 2 uniform ds_load_b128
#pragma unroll
        for (int i = 0; i < BPW; ++i) {
            const _Float16* wrow = wbase + (size_t)i * (WAVES * 16 * NPAD);
            v16h bb = frag16(wrow, kb, hiOff);
            acc[i] = __builtin_amdgcn_wmma_f32_16x16x32_f16(
                         false, a, false, bb, (short)0, acc[i], false, false);
        }
    }

    float acc_e = 0.0f;
#pragma unroll
    for (int i = 0; i < BPW; ++i) {
        int blk = wave + i * WAVES;
        if (blk < NBLK && lane < 16)          // padded blocks contribute 0
            acc_e += s_sum[blk * 16 + lane] * acc[i][0];   // row 0 of D = q[n]
    }

    float ssum = 0.0f;
    for (int n = tid; n < NNEUR; n += THREADS) ssum += s_sum[n];
    redA[tid] = acc_e; redB[tid] = ssum;
    __syncthreads();
    for (int s = THREADS / 2; s > 0; s >>= 1) {
        if (tid < s) { redA[tid] += redA[tid + s]; redB[tid] += redB[tid + s]; }
        __syncthreads();
    }
    float e = -0.5f * redA[0] + redB[0];
    __syncthreads();
    return e;
}

__global__ void __launch_bounds__(THREADS)
hopfield_kernel(const float* __restrict__ inp,
                float* __restrict__ Wf,
                _Float16* __restrict__ Wh,
                const int* __restrict__ thr_p,
                const int* __restrict__ niter_p,
                float* __restrict__ out) {
    // State as f16 in WMMA-A friendly row-major [16][NPAD]; rows 8..15 zero.
    __shared__ __align__(16) _Float16 st[16 * NPAD];   // 25.0 KB
    __shared__ __align__(16) _Float16 s16[NPAD];       // f16 shadow of s_sum
    __shared__ float s_sum[NPAD];                      // per-neuron sum over t
    __shared__ float in_cnt[NNEUR];                    // init-state sums (STDP)
    __shared__ float redA[THREADS];
    __shared__ float redB[THREADS];
    __shared__ int   sh_done;

    const int tid   = threadIdx.x;
    const int wave  = tid >> 5;
    const int lane  = tid & 31;
    const int nloc  = lane & 15;          // local neuron / row within fragment
    const int hiOff = (lane >> 4) * 8;    // K-half selector per ISA layout
    const float thr = (float)(*thr_p);
    const int niter = *niter_p;

    for (int b = 0; b < BATCH; ++b) {
        // ---------------- load initial state ----------------
        for (int i = tid; i < 16 * NPAD; i += THREADS) st[i] = (_Float16)0.0f;
        __syncthreads();
        const float* ib = inp + (size_t)b * TSTEPS * NNEUR;
        for (int i = tid; i < TSTEPS * NNEUR; i += THREADS) {
            int t = i / NNEUR, n = i - t * NNEUR;
            st[t * NPAD + n] = (_Float16)ib[i];
        }
        __syncthreads();
        for (int n = tid; n < NPAD; n += THREADS) {
            float s = 0.0f;
            if (n < NNEUR)
                for (int t = 0; t < TSTEPS; ++t) s += (float)st[t * NPAD + n];
            s_sum[n] = s;
            s16[n]   = (_Float16)s;
            if (n < NNEUR) in_cnt[n] = s;
        }
        __syncthreads();

        float e_prev = block_energy_wmma(Wh, s16, s_sum, redA, redB,
                                         tid, wave, lane, nloc, hiOff);
        if (tid == 0) sh_done = 0;

        // ---------------- relaxation iterations ----------------
        for (int it = 0; it < niter; ++it) {
            __syncthreads();
            if (sh_done) continue;                 // uniform branch

            // sweep: 7 groups x 8 waves, one 16-neuron block per wave
            for (int g = 0; g < 7; ++g) {
                int blk = g * WAVES + wave;
                v8f acc = {};
                if (blk < NBLK) {                  // wave-uniform guard
                    const _Float16* srow = st + nloc * NPAD;
                    const _Float16* wrow = Wh + (size_t)(blk * 16 + nloc) * NPAD;
                    for (int kc = 0; kc < KCHUNKS; ++kc) {
                        int kb = kc * 32;
                        __builtin_prefetch(wrow + ((kb + 64 < NPAD) ? kb + 64 : 0), 0, 1);
                        v16h a  = frag16(srow, kb, hiOff);  // ds_load_b128 x2
                        v16h bb = frag16(wrow, kb, hiOff);  // global_load_b128 x2
                        acc = __builtin_amdgcn_wmma_f32_16x16x32_f16(
                                  false, a, false, bb, (short)0, acc, false, false);
                    }
                }
                __syncthreads();   // potentials done; now apply updates
                if (blk < NBLK && lane < 16) {
                    // lanes 0..15 each own a full neuron column: acc[t]=pot(t)
                    int n = blk * 16 + lane;
                    float pot[TSTEPS];
#pragma unroll
                    for (int t = 0; t < TSTEPS; ++t) pot[t] = acc[t];
                    int t0 = TSTEPS;                       // first fire time
#pragma unroll
                    for (int t = TSTEPS - 1; t >= 0; --t)
                        if (pot[t] >= thr) t0 = t;
                    float cnt = 0.0f;
#pragma unroll
                    for (int t = 0; t < TSTEPS; ++t) {
                        float v = (t >= t0) ? 1.0f : 0.0f;
                        st[t * NPAD + n] = (_Float16)v;
                        cnt += v;
                    }
                    s_sum[n] = cnt;
                    s16[n]   = (_Float16)cnt;
                }
                __syncthreads();
            }

            float e_new = block_energy_wmma(Wh, s16, s_sum, redA, redB,
                                            tid, wave, lane, nloc, hiOff);
            if (tid == 0 && e_new == e_prev) sh_done = 1;
            e_prev = e_new;
        }
        __syncthreads();

        // ---------------- write output slice ----------------
        float* ob = out + (size_t)b * TSTEPS * NNEUR;
        for (int i = tid; i < TSTEPS * NNEUR; i += THREADS) {
            int t = i / NNEUR, n = i - t * NNEUR;
            ob[i] = (float)st[t * NPAD + n];
        }

        // ---------------- STDP update of W (f32 master + f16 mirror) -------
        unsigned bseed = hash_u32(0xC0FFEEu ^ ((unsigned)b * 0x9E3779B9u));
        for (int e = tid; e < NNEUR * NNEUR; e += THREADS) {
            int i = e / NNEUR;
            int j = e - i * NNEUR;
            float insp  = 8.0f - in_cnt[i];    // spike-latency codes (row bcast)
            float outsp = 8.0f - s_sum[i];
            bool inl = insp < 8.0f, outl = outsp < 8.0f;
            float w = Wf[e];
            float delta = 0.0f;
            if (inl || outl) {
                unsigned h0 = hash_u32(bseed ^ ((unsigned)e * 0x85EBCA6Bu));
                unsigned h1 = hash_u32(h0 + 0x68E31DA4u);
                unsigned h2 = hash_u32(h1 + 0xB5297A4Du);
                float u0 = u01(h0), u1 = u01(h1), u2 = u01(h2);
                float w8 = w * 0.125f;
                float um = (u2 < 0.1f) ? 1.0f : 0.0f;           // UMIN draw
                float ub = (u1 < 0.5f) ? 1.0f : 0.0f;           // branch draw
                if (inl && outl) {
                    if (insp <= outsp) {
                        float fp = (u0 < w8 * (2.0f - w8)) ? 1.0f : 0.0f;
                        delta =  ub * fmaxf(fp, um);            // capture
                    } else {
                        float fm = (u0 < (1.0f - w8) * (1.0f + w8)) ? 1.0f : 0.0f;
                        delta = -ub * fmaxf(fm, um);            // minus
                    }
                } else if (inl) {                                // search
                    float fp = (u0 < w8 * (2.0f - w8)) ? 1.0f : 0.0f;
                    delta =  ub * fmaxf(fp, um);
                } else {                                         // backoff
                    float fm = (u0 < (1.0f - w8) * (1.0f + w8)) ? 1.0f : 0.0f;
                    delta = -ub * fmaxf(fm, um);
                }
            }
            float wn = fminf(8.0f, fmaxf(0.0f, w + delta));
            Wf[e] = wn;
            Wh[(size_t)i * NPAD + j] = (_Float16)wn;
        }
        __syncthreads();   // make new W visible before next batch element
    }
}

extern "C" void kernel_launch(void* const* d_in, const int* in_sizes, int n_in,
                              void* d_out, int out_size, void* d_ws, size_t ws_size,
                              hipStream_t stream) {
    const float* inp    = (const float*)d_in[0];   // [64,8,784] spikes
    const float* W      = (const float*)d_in[1];   // [784,784]
    const int*   thr_p  = (const int*)d_in[2];     // scalar threshold
    const int*   nit_p  = (const int*)d_in[3];     // scalar num_iter
    float*       outp   = (float*)d_out;           // [64,8,28,28]

    char* ws = (char*)d_ws;
    _Float16* Wh = (_Float16*)ws;                               // 896*800*2 B
    float*    Wf = (float*)(ws + (size_t)NROWPAD * NPAD * 2);   // 784*784*4 B

    // Reset mutable weight copies every call (determinism requirement).
    hopfield_prep_kernel<<<(NROWPAD * NPAD + THREADS - 1) / THREADS, THREADS, 0,
                           stream>>>(W, Wf, Wh);
    // Entire sequential chain in one persistent workgroup.
    hopfield_kernel<<<1, THREADS, 0, stream>>>(inp, Wf, Wh, thr_p, nit_p, outp);
}
